// Job_encoder_85779086835959
// MI455X (gfx1250) — matrix operand order repaired and checked by reading, not compiled, not run on hardware
//
#include <hip/hip_runtime.h>
#include <hip/hip_bf16.h>

typedef _Float16 half_t;
typedef __attribute__((ext_vector_type(16))) _Float16 v16h;
typedef __attribute__((ext_vector_type(8)))  _Float16 v8h;
typedef __attribute__((ext_vector_type(8)))  float    v8f;

#define Bq 8
#define Nn 2048
#define Hh 64
#define NEG_BIG (-9.0e15f)
#define PST 72   // padded P-tile row stride in halfs (144B, 16B-aligned rows)

struct U128x2 { uint4 lo, hi; };
static __device__ __forceinline__ v16h make_frag(uint4 lo, uint4 hi) {
    U128x2 t{lo, hi};
    return __builtin_bit_cast(v16h, t);
}

// --------------------------------------------------------------------------
// h = x @ W ; fs = h . asrc ; fd = h . adst.
// h is emitted TRANSPOSED as f16: h16T[b][col][n]  (col-contiguous in n),
// so attention's WMMA B fragments are two contiguous 16B global loads per lane.
// Block = 256 threads -> 4 rows x 64 output channels.
// --------------------------------------------------------------------------
__global__ __launch_bounds__(256)
void gat_transform(const float* __restrict__ x, int Fin,
                   const float* __restrict__ W,
                   const float* __restrict__ asrc,
                   const float* __restrict__ adst,
                   half_t* __restrict__ h16T,
                   float* __restrict__ fs,
                   float* __restrict__ fd)
{
    __shared__ float sW[64 * 64];
    __shared__ float sx[4 * 64];
    __shared__ float red[4 * 64];
    const int tid  = threadIdx.x;
    const int row0 = blockIdx.x * 4;

    for (int i = tid; i < Fin * 64; i += 256) sW[i] = W[i];
    for (int i = tid; i < 4 * Fin; i += 256) {
        int rr = i / Fin, ff = i - rr * Fin;
        sx[rr * 64 + ff] = x[(size_t)(row0 + rr) * Fin + ff];
    }
    __syncthreads();

    const int r  = tid >> 6;
    const int hc = tid & 63;
    float acc = 0.f;
    for (int f = 0; f < Fin; ++f) acc += sx[r * 64 + f] * sW[f * 64 + hc];

    {   // transposed f16 store: h16T[(b*64 + hc)*N + n]
        int g = row0 + r;
        int bb = g >> 11;           // / Nn
        int n  = g & (Nn - 1);
        h16T[((size_t)bb * 64 + hc) * Nn + n] = (half_t)acc;
    }

    // fs reduction
    red[r * 64 + hc] = acc * asrc[hc];
    __syncthreads();
    for (int off = 32; off > 0; off >>= 1) {
        if (hc < off) red[r * 64 + hc] += red[r * 64 + hc + off];
        __syncthreads();
    }
    if (hc == 0) fs[row0 + r] = red[r * 64];
    __syncthreads();

    // fd reduction
    red[r * 64 + hc] = acc * adst[hc];
    __syncthreads();
    for (int off = 32; off > 0; off >>= 1) {
        if (hc < off) red[r * 64 + hc] += red[r * 64 + hc + off];
        __syncthreads();
    }
    if (hc == 0) fd[row0 + r] = red[r * 64];
}

// --------------------------------------------------------------------------
// Flash-style masked-softmax attention:
//   out[i,:] = tanh( (softmax_j(mask(lrelu(fs_i+fd_j))) @ h)[i,:] (+ res) )
// Block = 128 threads (4 waves) for 16 i-rows; j in tiles of 64 (2 WMMA-K).
// Scores/online-softmax fully in registers (width-8 shfl butterflies).
// P tile and rescale factors are DOUBLE-BUFFERED by tile parity ->
// exactly ONE barrier per 64-j tile. B fragments come straight from the
// transposed h in global memory (L2-resident).
// --------------------------------------------------------------------------
__global__ __launch_bounds__(128)
void gat_attention(const float* __restrict__ adj,
                   const half_t* __restrict__ h16T,   // [B][64][N]
                   const float* __restrict__ fs,
                   const float* __restrict__ fd,
                   const float* __restrict__ res,
                   float* __restrict__ outp)
{
    __shared__ __align__(16) half_t ph[2][16 * PST];  // double-buffered P tile
    __shared__ __align__(16) float scl[2][16];        // double-buffered rescale
    __shared__ float mrow[16], srow[16];

    const int tid  = threadIdx.x;
    const int lane = tid & 31;
    const int wave = tid >> 5;
    const int b    = blockIdx.x >> 7;                 // N/16 = 128 tiles/batch
    const int i0   = (blockIdx.x & 127) << 4;
    const int row  = tid >> 3;                        // 16 rows x 8 lanes
    const int l8   = tid & 7;
    const int hbase = wave << 4;

    const size_t adjRow = ((size_t)b * Nn + i0 + row) * Nn;
    const float* fdB = &fd[b * Nn];
    const float fsv = fs[b * Nn + i0 + row];

    if (tid < 16) { mrow[tid] = -3.0e38f; srow[tid] = 0.f; }
    __syncthreads();

    v8f acc = {};

    // fragment addressing constants
    const int arow = lane & 15;
    const int kh   = lane >> 4;                       // 0: lanes 0-15, 1: 16-31
    const uint4* prA0 = (const uint4*)&ph[0][arow * PST];
    const uint4* prA1 = (const uint4*)&ph[1][arow * PST];
    const int bcol = hbase + (lane & 15);
    const half_t* hcolp = &h16T[((size_t)b * 64 + bcol) * Nn + (kh << 3)];
    const int roff = kh << 3;

    for (int j0 = 0; j0 < Nn; j0 += 64) {
        const int par = (j0 >> 6) & 1;

        // ---- scores for 8 consecutive j, all in registers ----
        float ev[8];
        {
            const float* ap = &adj[adjRow + j0 + l8 * 8];
            float4 a0 = *(const float4*)ap;
            float4 a1 = *(const float4*)(ap + 4);
            const float* fp = &fdB[j0 + l8 * 8];
            float4 f0 = *(const float4*)fp;
            float4 f1 = *(const float4*)(fp + 4);
            float aa[8] = {a0.x, a0.y, a0.z, a0.w, a1.x, a1.y, a1.z, a1.w};
            float ff[8] = {f0.x, f0.y, f0.z, f0.w, f1.x, f1.y, f1.z, f1.w};
            #pragma unroll
            for (int t = 0; t < 8; ++t) {
                float e = fsv + ff[t];
                e = fmaxf(e, 0.2f * e);               // exact leaky-relu, one max
                ev[t] = (aa[t] > 0.f) ? e : NEG_BIG;
            }
        }

        // ---- online softmax stats (width-8 butterflies -> all lanes) ----
        float pm = ev[0];
        #pragma unroll
        for (int t = 1; t < 8; ++t) pm = fmaxf(pm, ev[t]);
        pm = fmaxf(pm, __shfl_xor(pm, 4, 8));
        pm = fmaxf(pm, __shfl_xor(pm, 2, 8));
        pm = fmaxf(pm, __shfl_xor(pm, 1, 8));
        const float mo = mrow[row];                   // same-wave read-before-write
        const float mn = fmaxf(mo, pm);
        const float rs = __expf(mo - mn);
        if (l8 == 0) { mrow[row] = mn; scl[par][row] = rs; srow[row] *= rs; }

        // ---- P = exp(e - m), packed f16, one b128 LDS store ----
        v8h pp;
        float ps = 0.f;
        #pragma unroll
        for (int t = 0; t < 8; ++t) {
            float p = __expf(ev[t] - mn);
            pp[t] = (half_t)p;
            ps += p;
        }
        *(v8h*)&ph[par][row * PST + l8 * 8] = pp;
        ps += __shfl_xor(ps, 4, 8);
        ps += __shfl_xor(ps, 2, 8);
        ps += __shfl_xor(ps, 1, 8);
        if (l8 == 0) srow[row] += ps;

        __syncthreads();    // single barrier: P tile + scl[par] visible

        // ---- rescale f32 accumulator (two b128 LDS loads) ----
        {
            float4 s0 = *(const float4*)&scl[par][roff];
            float4 s1 = *(const float4*)&scl[par][roff + 4];
            acc[0] *= s0.x; acc[1] *= s0.y; acc[2] *= s0.z; acc[3] *= s0.w;
            acc[4] *= s1.x; acc[5] *= s1.y; acc[6] *= s1.z; acc[7] *= s1.w;
        }

        // ---- two WMMAs: K = j0..j0+31 and j0+32..j0+63 ----
        {
            const uint4* prA = par ? prA1 : prA0;
            v16h af0 = make_frag(prA[kh],     prA[2 + kh]);
            v16h af1 = make_frag(prA[4 + kh], prA[6 + kh]);
            const half_t* hp = hcolp + j0;
            v16h bf0 = make_frag(*(const uint4*)hp,        *(const uint4*)(hp + 16));
            v16h bf1 = make_frag(*(const uint4*)(hp + 32), *(const uint4*)(hp + 48));
            acc = __builtin_amdgcn_wmma_f32_16x16x32_f16(
                false, af0, false, bf0, (short)0, acc, false, false);
            acc = __builtin_amdgcn_wmma_f32_16x16x32_f16(
                false, af1, false, bf1, (short)0, acc, false, false);
        }
        // no trailing barrier: buffer `par` is not rewritten until two tiles
        // later, and the next tile's barrier orders those writes after all
        // reads of this tile's buffer.
    }
    __syncthreads();        // srow final values visible (defensive)

    // ---- normalize, residual, tanh, store (C/D lane layout) ----
    {
        int rro = (lane >> 4) << 3;
        int col = hbase + (lane & 15);
        #pragma unroll
        for (int r = 0; r < 8; ++r) {
            int rw = r + rro;
            float v = acc[r] / srow[rw];
            size_t gi = ((size_t)(b * Nn + i0 + rw)) * 64 + col;
            if (res) v += res[gi];
            outp[gi] = tanhf(v);
        }
    }
}

// --------------------------------------------------------------------------
// out[b,h] = mean_n t[b,n,h]
// --------------------------------------------------------------------------
__global__ __launch_bounds__(64)
void gat_mean(const float* __restrict__ t, float* __restrict__ outp)
{
    int b = blockIdx.x, h = threadIdx.x;
    float s = 0.f;
    for (int n = 0; n < Nn; ++n) s += t[((size_t)(b * Nn + n)) * 64 + h];
    outp[b * 64 + h] = s * (1.0f / Nn);
}

extern "C" void kernel_launch(void* const* d_in, const int* in_sizes, int n_in,
                              void* d_out, int out_size, void* d_ws, size_t ws_size,
                              hipStream_t stream)
{
    const float* adj  = (const float*)d_in[0];
    const float* feat = (const float*)d_in[1];
    // d_in[2] = mask (unused by the reference computation)
    const float* W1 = (const float*)d_in[3];
    const float* W2 = (const float*)d_in[4];
    const float* W3 = (const float*)d_in[5];
    const float* W4 = (const float*)d_in[6];
    const float* as1 = (const float*)d_in[7];
    const float* ad1 = (const float*)d_in[8];
    const float* as2 = (const float*)d_in[9];
    const float* ad2 = (const float*)d_in[10];
    const float* as3 = (const float*)d_in[11];
    const float* ad3 = (const float*)d_in[12];
    const float* as4 = (const float*)d_in[13];
    const float* ad4 = (const float*)d_in[14];

    const int BN = Bq * Nn;
    char* ws = (char*)d_ws;
    half_t* h16T = (half_t*)ws;  ws += (size_t)BN * 64 * 2;
    float*  fs   = (float*)ws;   ws += (size_t)BN * 4;
    float*  fd   = (float*)ws;   ws += (size_t)BN * 4;
    float*  tA   = (float*)ws;   ws += (size_t)BN * 64 * 4;
    float*  tB   = (float*)ws;

    dim3 gT(BN / 4), bT(256);
    dim3 gA(Bq * (Nn / 16)), bA(128);

    // layer 1: t1 = tanh(gat(feature))
    gat_transform<<<gT, bT, 0, stream>>>(feat, 16, W1, as1, ad1, h16T, fs, fd);
    gat_attention<<<gA, bA, 0, stream>>>(adj, h16T, fs, fd, nullptr, tA);
    // layer 2: t2 = tanh(gat(t1) + t1)
    gat_transform<<<gT, bT, 0, stream>>>(tA, 64, W2, as2, ad2, h16T, fs, fd);
    gat_attention<<<gA, bA, 0, stream>>>(adj, h16T, fs, fd, tA, tB);
    // layer 3: t3 = tanh(gat(t2) + t2)
    gat_transform<<<gT, bT, 0, stream>>>(tB, 64, W3, as3, ad3, h16T, fs, fd);
    gat_attention<<<gA, bA, 0, stream>>>(adj, h16T, fs, fd, tB, tA);
    // layer 4: t4 = tanh(t3 + gat(t3))
    gat_transform<<<gT, bT, 0, stream>>>(tA, 64, W4, as4, ad4, h16T, fs, fd);
    gat_attention<<<gA, bA, 0, stream>>>(adj, h16T, fs, fd, tA, tB);
    // mean over nodes -> [B,H]
    gat_mean<<<dim3(Bq), dim3(64), 0, stream>>>(tB, (float*)d_out);
}
